// RWKV_59261958750757
// MI455X (gfx1250) — matrix operand order repaired and checked by reading, not compiled, not run on hardware
//
#include <hip/hip_runtime.h>
#include <hip/hip_bf16.h>
#include <stdint.h>

// RWKV-5 chunked forward for MI455X (gfx1250, wave32, WMMA bf16 16x16x32).
// B=4 T=2048 C=1024 H=16 N=64 Q=256 nc=8

#define BB 4
#define TT 2048
#define CC 1024
#define HH 16
#define NN 64
#define QQ 256
#define NC 8

typedef __attribute__((ext_vector_type(16))) __bf16        bf16x16;
typedef __attribute__((ext_vector_type(8)))  float         f32x8;
typedef __attribute__((ext_vector_type(4)))  float         f32x4;
typedef __attribute__((ext_vector_type(4)))  unsigned int  u32x4;

union FragU {
  u32x4 q[2];
  bf16x16 bf;
  unsigned short s[16];
};

__device__ __forceinline__ unsigned short f2bf(float f) {
  unsigned int u = __float_as_uint(f);
  u += 0x7fffu + ((u >> 16) & 1u);          // round-to-nearest-even
  return (unsigned short)(u >> 16);
}
__device__ __forceinline__ float bf2f(unsigned short s) {
  return __uint_as_float(((unsigned int)s) << 16);
}
__device__ __forceinline__ unsigned int pack2(float a, float b) {
  return (unsigned int)f2bf(a) | ((unsigned int)f2bf(b) << 16);
}

__device__ __forceinline__ f32x8 wmma_bf16(bf16x16 a, bf16x16 b, f32x8 c) {
  // (neg_a, A, neg_b, B, c_mod, C, reuse_a, reuse_b)
  return __builtin_amdgcn_wmma_f32_16x16x32_bf16(false, a, false, b, (short)0, c,
                                                 false, false);
}

// A-matrix fragment (16x32 bf16, MxK), memory row-major [M][K] with `stride` elems.
// ISA 7.12.2: lanes 0-15 M=lane, lanes 16-31 same M rows; per half-wave the K runs
// are {kBase+8*half .. +7} (VGPR0-3) and {kBase+16+8*half .. +7} (VGPR4-7).
__device__ __forceinline__ bf16x16 ld_afrag(const unsigned short* base, int stride,
                                            int row, int kBase, int lane) {
  int m = lane & 15, half = lane >> 4;
  const unsigned short* p = base + (size_t)(row + m) * (size_t)stride + kBase + 8 * half;
  FragU u;
  u.q[0] = *(const u32x4*)p;
  u.q[1] = *(const u32x4*)(p + 16);
  return u.bf;
}

// B-matrix fragment (32x16 bf16, KxN), memory layout [N][K] (i.e. B transposed =
// natural layout for W row-major / k row-major / vT / ST). Lane n = lane&15,
// K runs = {kBase+8*half .. +7} and {kBase+16+8*half .. +7} (matches A fragment).
__device__ __forceinline__ bf16x16 ld_bfrag(const unsigned short* base, int stride,
                                            int col, int kBase, int lane) {
  int n = lane & 15, half = lane >> 4;
  const unsigned short* p = base + (size_t)(col + n) * (size_t)stride + kBase + 8 * half;
  FragU u;
  u.q[0] = *(const u32x4*)p;
  u.q[1] = *(const u32x4*)(p + 16);
  return u.bf;
}

// A fragment with per-K scaling (for (k * wk)^T), wk indexed by chunk-local q.
__device__ __forceinline__ bf16x16 ld_afrag_scaled(const unsigned short* base, int stride,
                                                   int row, int kBase, int lane,
                                                   const float* wk) {
  int m = lane & 15, half = lane >> 4;
  const unsigned short* p = base + (size_t)(row + m) * (size_t)stride;
  FragU u;
#pragma unroll
  for (int e = 0; e < 8; ++e) {
    int q0 = kBase + 8 * half + e;
    int q1 = kBase + 16 + 8 * half + e;
    u.s[e]     = f2bf(bf2f(p[q0]) * wk[q0]);
    u.s[e + 8] = f2bf(bf2f(p[q1]) * wk[q1]);
  }
  return u.bf;
}

// ---------------------------------------------------------------------------
// Kernel 1: fused token-shift mixing + projection  (r/k/v, launched 3x)
// out[m][n] = sum_k mix(x)[m][k] * W[n][k] + bias[n]
// WRITE_ROW: bf16 [B,H,T,N]; WRITE_COL: bf16 [B,H,N,T] (compile-time selection)
// ---------------------------------------------------------------------------
template <bool WRITE_ROW, bool WRITE_COL>
__global__ void __launch_bounds__(256)
gemm_mix_kernel(const float* __restrict__ x, const float* __restrict__ maa,
                const float* __restrict__ W, const float* __restrict__ bias,
                unsigned short* __restrict__ dstRow, unsigned short* __restrict__ dstColT) {
  __shared__ __align__(16) unsigned short As[128 * 32];
  __shared__ __align__(16) unsigned short Bs[128 * 32];

  const int tid = threadIdx.x, lane = tid & 31, w = tid >> 5;
  const int wm = w & 3, wn = w >> 2;           // 4 M-strips x 2 N-strips
  const int gm0 = blockIdx.x * 128;
  const int gn0 = blockIdx.y * 128;

  f32x8 acc[2][4] = {};

  const int arow = tid >> 1;                   // 0..127
  const int aseg = (tid & 1) * 16;             // k sub-offset 0/16
  const int grow = gm0 + arow;                 // global M row = b*T + t
  const bool first = ((grow & (TT - 1)) == 0);
  const float* xr = x + (size_t)grow * CC;
  // Branchless shifted row: always-valid address, zeroed by scale factor.
  const float* xp = first ? xr : (xr - CC);
  const float pscale = first ? 0.0f : 1.0f;
  const float* wrow = W + (size_t)(gn0 + arow) * CC;

  for (int k0 = 0; k0 < CC; k0 += 32) {
    __syncthreads();
    // --- stage A: mix + f32->bf16 (vector b128 loads, branchless shift) ---
    {
      const int kk = k0 + aseg;
      __builtin_prefetch((const void*)(xr + kk + 32), 0, 1);
      f32x4 xv[4], pv[4], mv[4];
#pragma unroll
      for (int s = 0; s < 4; ++s) {
        xv[s] = *(const f32x4*)(xr + kk + 4 * s);
        pv[s] = *(const f32x4*)(xp + kk + 4 * s);
        mv[s] = *(const f32x4*)(maa + kk + 4 * s);
      }
      unsigned int packed[8];
#pragma unroll
      for (int s = 0; s < 4; ++s) {
        f32x4 mixv = xv[s] + (pv[s] * pscale - xv[s]) * mv[s];
        packed[2 * s]     = pack2(mixv[0], mixv[1]);
        packed[2 * s + 1] = pack2(mixv[2], mixv[3]);
      }
      u32x4* d = (u32x4*)&As[arow * 32 + aseg];
      d[0] = (u32x4){packed[0], packed[1], packed[2], packed[3]};
      d[1] = (u32x4){packed[4], packed[5], packed[6], packed[7]};
    }
    // --- stage B: W f32->bf16 (vector b128 loads) ---
    {
      const float* wp = wrow + k0 + aseg;
      __builtin_prefetch((const void*)(wp + 32), 0, 1);
      unsigned int packed[8];
#pragma unroll
      for (int s = 0; s < 4; ++s) {
        f32x4 wv = *(const f32x4*)(wp + 4 * s);
        packed[2 * s]     = pack2(wv[0], wv[1]);
        packed[2 * s + 1] = pack2(wv[2], wv[3]);
      }
      u32x4* d = (u32x4*)&Bs[arow * 32 + aseg];
      d[0] = (u32x4){packed[0], packed[1], packed[2], packed[3]};
      d[1] = (u32x4){packed[4], packed[5], packed[6], packed[7]};
    }
    __syncthreads();

    bf16x16 bfr[4];
#pragma unroll
    for (int nt = 0; nt < 4; ++nt)
      bfr[nt] = ld_bfrag(Bs, 32, wn * 64 + nt * 16, 0, lane);
#pragma unroll
    for (int rt = 0; rt < 2; ++rt) {
      bf16x16 af = ld_afrag(As, 32, wm * 32 + rt * 16, 0, lane);
#pragma unroll
      for (int nt = 0; nt < 4; ++nt)
        acc[rt][nt] = wmma_bf16(af, bfr[nt], acc[rt][nt]);
    }
  }

  // --- epilogue: +bias, write bf16 (row-major and/or transposed) ---
  const int half = lane >> 4, nlane = lane & 15;
#pragma unroll
  for (int rt = 0; rt < 2; ++rt)
#pragma unroll
    for (int nt = 0; nt < 4; ++nt) {
      int gc = gn0 + wn * 64 + nt * 16 + nlane;
      float bia = bias[gc];
      int h = gc >> 6, nidx = gc & 63;
#pragma unroll
      for (int e = 0; e < 8; ++e) {
        int gr = gm0 + wm * 32 + rt * 16 + e + 8 * half;  // b*T + t
        int b = gr >> 11, t = gr & (TT - 1);
        unsigned short v16 = f2bf(acc[rt][nt][e] + bia);
        if (WRITE_ROW)
          dstRow[(((size_t)(b * HH + h)) * TT + t) * NN + nidx] = v16;
        if (WRITE_COL)
          dstColT[(((size_t)(b * HH + h)) * NN + nidx) * TT + t] = v16;
      }
    }
}

// ---------------------------------------------------------------------------
// Kernel 2: per-chunk intra attention + chunk KV.
// grid = B*H*nc; block = 256 (8 waves); each wave owns 32 rows of Q.
// ---------------------------------------------------------------------------
__global__ void __launch_bounds__(256)
wkv_inner_kernel(const unsigned short* __restrict__ r, const unsigned short* __restrict__ k,
                 const unsigned short* __restrict__ kT, const unsigned short* __restrict__ vT,
                 const float* __restrict__ time_decay, const float* __restrict__ time_faaaa,
                 float* __restrict__ yintra, float* __restrict__ kvbuf) {
  __shared__ __align__(16) unsigned short Astrip[8][16 * 32];
  __shared__ float wkArr[QQ];

  const int tid = threadIdx.x, lane = tid & 31, w = tid >> 5;
  const int bid = blockIdx.x;
  const int c = bid & 7, h = (bid >> 3) & 15, b = bid >> 7;
  const float ed = __expf(time_decay[h]);     // wd = exp(-ed); wd^e = exp(-e*ed)
  const float u  = time_faaaa[h];
  if (tid < QQ) wkArr[tid] = __expf(-(float)(QQ - 1 - tid) * ed);
  __syncthreads();

  const size_t bh = (size_t)(b * HH + h);
  const unsigned short* rC  = r  + (bh * TT + (size_t)c * QQ) * NN;
  const unsigned short* kC  = k  + (bh * TT + (size_t)c * QQ) * NN;
  const unsigned short* kTC = kT + bh * NN * TT + (size_t)c * QQ;   // stride TT
  const unsigned short* vTC = vT + bh * NN * TT + (size_t)c * QQ;   // stride TT

  const int m0 = w * 32;
  const int half = lane >> 4, nlane = lane & 15;
  f32x8 yacc[2][4] = {};

  for (int j0 = 0; j0 < QQ; j0 += 32) {
    bf16x16 vf[4];
#pragma unroll
    for (int nt = 0; nt < 4; ++nt)
      vf[nt] = ld_bfrag(vTC, TT, nt * 16, j0, lane);

#pragma unroll
    for (int rt = 0; rt < 2; ++rt) {
      // A(16x32 strip) = (r . k^T) o Wmat, in bf16 via wave-private LDS strip
#pragma unroll
      for (int jt = 0; jt < 2; ++jt) {
        f32x8 a = {};
#pragma unroll
        for (int ks = 0; ks < NN; ks += 32) {
          bf16x16 af = ld_afrag(rC, NN, m0 + rt * 16, ks, lane);
          bf16x16 bf = ld_bfrag(kC, NN, j0 + jt * 16, ks, lane);
          a = wmma_bf16(af, bf, a);
        }
#pragma unroll
        for (int e = 0; e < 8; ++e) {
          int i = m0 + rt * 16 + e + 8 * half;
          int j = j0 + jt * 16 + nlane;
          // Branchless Wmat factor: exp computed unconditionally, selected by
          // two v_cndmask (out-of-range exp value is never used).
          float expv = __expf(-(float)(i - j - 1) * ed);
          float fac = (i > j) ? expv : 0.0f;
          fac = (i == j) ? u : fac;
          Astrip[w][(e + 8 * half) * 32 + jt * 16 + nlane] = f2bf(a[e] * fac);
        }
      }
      // y_intra += A . v  (B fragments from vT)
      bf16x16 afrag = ld_afrag(Astrip[w], 32, 0, 0, lane);
#pragma unroll
      for (int nt = 0; nt < 4; ++nt)
        yacc[rt][nt] = wmma_bf16(afrag, vf[nt], yacc[rt][nt]);
    }
  }

  float* yC = yintra + (size_t)bid * QQ * NN;  // == [b,h, c*Q+q, n]
#pragma unroll
  for (int rt = 0; rt < 2; ++rt)
#pragma unroll
    for (int nt = 0; nt < 4; ++nt)
#pragma unroll
      for (int e = 0; e < 8; ++e) {
        int q = m0 + rt * 16 + e + 8 * half;
        int n = nt * 16 + nlane;
        yC[(size_t)q * NN + n] = yacc[rt][nt][e];
      }

  // KV_c = (k o wk)^T . v   (64x64, 16 tiles, 2 per wave)
  float* kvC = kvbuf + (size_t)bid * (NN * NN);
#pragma unroll
  for (int tloc = 0; tloc < 2; ++tloc) {
    int tt = w * 2 + tloc;
    int ti = tt >> 2, tj = tt & 3;
    f32x8 kv = {};
    for (int ks = 0; ks < QQ; ks += 32) {
      bf16x16 af = ld_afrag_scaled(kTC, TT, ti * 16, ks, lane, wkArr);
      bf16x16 bf = ld_bfrag(vTC, TT, tj * 16, ks, lane);
      kv = wmma_bf16(af, bf, kv);
    }
#pragma unroll
    for (int e = 0; e < 8; ++e) {
      int ni = ti * 16 + e + 8 * half;
      int nj = tj * 16 + nlane;
      kvC[ni * NN + nj] = kv[e];
    }
  }
}

// ---------------------------------------------------------------------------
// Kernel 3: sequential state scan over chunks (tiny). grid = B*H.
// Emits per-chunk incoming state, transposed, bf16 (B-fragment layout) and
// the final f32 state to d_out.
// ---------------------------------------------------------------------------
__global__ void __launch_bounds__(256)
state_scan_kernel(const float* __restrict__ kvbuf, const float* __restrict__ time_decay,
                  unsigned short* __restrict__ STbuf, float* __restrict__ state_out) {
  const int bid = blockIdx.x;                  // b*H + h
  const int h = bid & 15;
  const int tid = threadIdx.x;
  const float ed = __expf(time_decay[h]);
  const float wsc = __expf(-(float)QQ * ed);   // wd^Q
  float S[16];
#pragma unroll
  for (int e = 0; e < 16; ++e) S[e] = 0.0f;
  const int base = tid * 16;
  for (int c = 0; c < NC; ++c) {
    const size_t off = ((size_t)bid * NC + c) * (NN * NN);
    unsigned short* st = STbuf + off;
#pragma unroll
    for (int e = 0; e < 16; ++e) {             // incoming state, transposed
      int idx = base + e, i = idx >> 6, j = idx & 63;
      st[j * NN + i] = f2bf(S[e]);
    }
    const float* kv = kvbuf + off;
#pragma unroll
    for (int e = 0; e < 16; ++e) S[e] = wsc * S[e] + kv[base + e];
  }
  float* so = state_out + (size_t)bid * (NN * NN);
#pragma unroll
  for (int e = 0; e < 16; ++e) so[base + e] = S[e];
}

// ---------------------------------------------------------------------------
// Kernel 4: y = y_intra + (r . S_c) * wb[q], written bf16 row-major [B*T, C].
// grid = B*H*nc.
// ---------------------------------------------------------------------------
__global__ void __launch_bounds__(256)
y_combine_kernel(const unsigned short* __restrict__ r, const unsigned short* __restrict__ STbuf,
                 const float* __restrict__ yintra, const float* __restrict__ time_decay,
                 unsigned short* __restrict__ ybuf) {
  const int tid = threadIdx.x, lane = tid & 31, w = tid >> 5;
  const int bid = blockIdx.x;
  const int c = bid & 7, h = (bid >> 3) & 15, b = bid >> 7;
  const float ed = __expf(time_decay[h]);
  const size_t bh = (size_t)(b * HH + h);
  const unsigned short* rC  = r + (bh * TT + (size_t)c * QQ) * NN;
  const unsigned short* stC = STbuf + (size_t)bid * (NN * NN);   // S^T [n_out][n_in]
  const float* yC = yintra + (size_t)bid * QQ * NN;
  const int m0 = w * 32;
  const int half = lane >> 4, nlane = lane & 15;

  f32x8 cacc[2][4] = {};
#pragma unroll
  for (int ks = 0; ks < NN; ks += 32) {
    bf16x16 bfr[4];
#pragma unroll
    for (int nt = 0; nt < 4; ++nt)
      bfr[nt] = ld_bfrag(stC, NN, nt * 16, ks, lane);
#pragma unroll
    for (int rt = 0; rt < 2; ++rt) {
      bf16x16 af = ld_afrag(rC, NN, m0 + rt * 16, ks, lane);
#pragma unroll
      for (int nt = 0; nt < 4; ++nt)
        cacc[rt][nt] = wmma_bf16(af, bfr[nt], cacc[rt][nt]);
    }
  }

#pragma unroll
  for (int rt = 0; rt < 2; ++rt)
#pragma unroll
    for (int nt = 0; nt < 4; ++nt)
#pragma unroll
      for (int e = 0; e < 8; ++e) {
        int q = m0 + rt * 16 + e + 8 * half;
        int n = nt * 16 + nlane;
        float wb = __expf(-(float)q * ed);     // wd^q, chunk-local
        float y = yC[(size_t)q * NN + n] + cacc[rt][nt][e] * wb;
        int t = c * QQ + q;
        ybuf[((size_t)b * TT + t) * CC + h * NN + n] = f2bf(y);
      }
}

// ---------------------------------------------------------------------------
// Kernel 5: out = y . Wo^T + bo  (A already bf16)
// ---------------------------------------------------------------------------
__global__ void __launch_bounds__(256)
gemm_out_kernel(const unsigned short* __restrict__ ybuf, const float* __restrict__ W,
                const float* __restrict__ bias, float* __restrict__ out) {
  __shared__ __align__(16) unsigned short As[128 * 32];
  __shared__ __align__(16) unsigned short Bs[128 * 32];

  const int tid = threadIdx.x, lane = tid & 31, w = tid >> 5;
  const int wm = w & 3, wn = w >> 2;
  const int gm0 = blockIdx.x * 128;
  const int gn0 = blockIdx.y * 128;

  f32x8 acc[2][4] = {};
  const int arow = tid >> 1;
  const int aseg = (tid & 1) * 16;
  const unsigned short* arp = ybuf + (size_t)(gm0 + arow) * CC;
  const float* wrow = W + (size_t)(gn0 + arow) * CC;

  for (int k0 = 0; k0 < CC; k0 += 32) {
    __syncthreads();
    {
      const unsigned short* ap = arp + k0 + aseg;
      __builtin_prefetch((const void*)(ap + 32), 0, 1);
      u32x4 a0 = *(const u32x4*)ap;
      u32x4 a1 = *(const u32x4*)(ap + 8);
      u32x4* d = (u32x4*)&As[arow * 32 + aseg];
      d[0] = a0;
      d[1] = a1;
    }
    {
      const float* wp = wrow + k0 + aseg;
      __builtin_prefetch((const void*)(wp + 32), 0, 1);
      unsigned int packed[8];
#pragma unroll
      for (int s = 0; s < 4; ++s) {
        f32x4 wv = *(const f32x4*)(wp + 4 * s);
        packed[2 * s]     = pack2(wv[0], wv[1]);
        packed[2 * s + 1] = pack2(wv[2], wv[3]);
      }
      u32x4* d = (u32x4*)&Bs[arow * 32 + aseg];
      d[0] = (u32x4){packed[0], packed[1], packed[2], packed[3]};
      d[1] = (u32x4){packed[4], packed[5], packed[6], packed[7]};
    }
    __syncthreads();

    bf16x16 bfr[4];
#pragma unroll
    for (int nt = 0; nt < 4; ++nt)
      bfr[nt] = ld_bfrag(Bs, 32, wn * 64 + nt * 16, 0, lane);
#pragma unroll
    for (int rt = 0; rt < 2; ++rt) {
      bf16x16 af = ld_afrag(As, 32, wm * 32 + rt * 16, 0, lane);
#pragma unroll
      for (int nt = 0; nt < 4; ++nt)
        acc[rt][nt] = wmma_bf16(af, bfr[nt], acc[rt][nt]);
    }
  }

  const int half = lane >> 4, nlane = lane & 15;
#pragma unroll
  for (int rt = 0; rt < 2; ++rt)
#pragma unroll
    for (int nt = 0; nt < 4; ++nt) {
      int gc = gn0 + wn * 64 + nt * 16 + nlane;
      float bia = bias[gc];
#pragma unroll
      for (int e = 0; e < 8; ++e) {
        int gr = gm0 + wm * 32 + rt * 16 + e + 8 * half;
        out[(size_t)gr * CC + gc] = acc[rt][nt][e] + bia;
      }
    }
}

// ---------------------------------------------------------------------------
extern "C" void kernel_launch(void* const* d_in, const int* in_sizes, int n_in,
                              void* d_out, int out_size, void* d_ws, size_t ws_size,
                              hipStream_t stream) {
  const float* x     = (const float*)d_in[0];
  const float* maa_k = (const float*)d_in[1];
  const float* maa_v = (const float*)d_in[2];
  const float* maa_r = (const float*)d_in[3];
  const float* tdec  = (const float*)d_in[4];
  const float* tfaa  = (const float*)d_in[5];
  const float* Wr = (const float*)d_in[6];
  const float* br = (const float*)d_in[7];
  const float* Wk = (const float*)d_in[8];
  const float* bk = (const float*)d_in[9];
  const float* Wv = (const float*)d_in[10];
  const float* bv = (const float*)d_in[11];
  const float* Wo = (const float*)d_in[12];
  const float* bo = (const float*)d_in[13];
  float* out = (float*)d_out;

  const size_t M = (size_t)BB * TT;            // 8192
  char* ws = (char*)d_ws;
  size_t off = 0;
  unsigned short* rB  = (unsigned short*)(ws + off); off += M * CC * 2;   // 16MB
  unsigned short* kB  = (unsigned short*)(ws + off); off += M * CC * 2;   // 16MB
  unsigned short* kT  = (unsigned short*)(ws + off); off += M * CC * 2;   // 16MB
  unsigned short* vT  = (unsigned short*)(ws + off); off += M * CC * 2;   // 16MB
  float* yintra       = (float*)(ws + off);          off += M * CC * 4;   // 32MB
  float* kvbuf        = (float*)(ws + off);          off += (size_t)BB * HH * NC * NN * NN * 4; // 8MB
  unsigned short* STb = (unsigned short*)(ws + off); off += (size_t)BB * HH * NC * NN * NN * 2; // 4MB
  unsigned short* yB  = (unsigned short*)(ws + off); off += M * CC * 2;   // 16MB

  dim3 gg(64, 8), tb(256);
  gemm_mix_kernel<true,  false><<<gg, tb, 0, stream>>>(x, maa_r, Wr, br, rB, nullptr);
  gemm_mix_kernel<true,  true ><<<gg, tb, 0, stream>>>(x, maa_k, Wk, bk, kB, kT);
  gemm_mix_kernel<false, true ><<<gg, tb, 0, stream>>>(x, maa_v, Wv, bv, nullptr, vT);

  wkv_inner_kernel<<<BB * HH * NC, 256, 0, stream>>>(rB, kB, kT, vT, tdec, tfaa,
                                                     yintra, kvbuf);
  state_scan_kernel<<<BB * HH, 256, 0, stream>>>(kvbuf, tdec, STb,
                                                 out + (size_t)BB * TT * CC);
  y_combine_kernel<<<BB * HH * NC, 256, 0, stream>>>(rB, STb, yintra, tdec, yB);

  gemm_out_kernel<<<gg, tb, 0, stream>>>(yB, Wo, bo, out);
}
